// PairwiseScore_13091060318543
// MI455X (gfx1250) — matrix operand order repaired and checked by reading, not compiled, not run on hardware
//
#include <hip/hip_runtime.h>

// ---------------------------------------------------------------------------
// Types for CDNA5 WMMA (wave32, 16x16x32 bf16 -> f32)
// ---------------------------------------------------------------------------
typedef __bf16 v16bf __attribute__((ext_vector_type(16)));
typedef __bf16 v8bf  __attribute__((ext_vector_type(8)));
typedef float  v8f   __attribute__((ext_vector_type(8)));

// Problem constants (match reference setup_inputs)
#define GDIM   1200
#define GPAD   1216      // 38 * 32
#define PNUM   2000
#define HID    150
#define HPAD   160       // 10 * 16
#define KWIN   50
// workspace byte offsets (all 32B aligned)
#define OFF_G      0u            // f32 [2000][1216]           9,728,000 B
#define OFF_AB     9728000u      // f32 [2000][320]            2,560,000 B
#define OFF_W1ABT  12288000u     // bf16 [320][1216]             778,240 B
#define OFF_W1CT   13066240u     // bf16 [160][1216]             389,120 B
#define OFF_W2T    13455360u     // bf16 [160][160]               51,200 B

__device__ __forceinline__ int imin(int a, int b) { return a < b ? a : b; }

// f32 -> bf16 (round to nearest even), as raw u16
__device__ __forceinline__ unsigned short f2bf(float f) {
    unsigned int u = __builtin_bit_cast(unsigned int, f);
    u += 0x7FFFu + ((u >> 16) & 1u);
    return (unsigned short)(u >> 16);
}

// A-fragment (16x32 bf16) from an LDS row-major u16 array with given stride.
// Layout: lane L<16 holds row M=L, K=0..15 contiguous; lanes 16..31 hold same
// rows, K=16..31 (matches B-fragment order).
__device__ __forceinline__ v16bf load_a_lds(const unsigned short* s, int stride,
                                            int mbase, int k0, int lane) {
    int m    = mbase + (lane & 15);
    int half = lane >> 4;
    const unsigned short* p = s + m * stride + k0 + half * 16;
    v8bf lo = *reinterpret_cast<const v8bf*>(p);        // 16B: K chunk A
    v8bf hi = *reinterpret_cast<const v8bf*>(p + 8);    // 16B: K chunk B
    return __builtin_shufflevector(lo, hi, 0, 1, 2, 3, 4, 5, 6, 7,
                                           8, 9, 10, 11, 12, 13, 14, 15);
}

// B-fragment (32x16 bf16) from a column-major (W^T row-major) u16 array.
// Layout used here (contiguous k): lane n<16 holds column N=n, K=0..15 contiguous; lanes 16..31
// hold K=16..31. One contiguous 32-byte load per lane.
__device__ __forceinline__ v16bf load_b_glb(const unsigned short* wt, int ldk,
                                            int col, int k0, int lane) {
    const unsigned short* p = wt + col * ldk + k0 + (lane >> 4) * 16;
    return *reinterpret_cast<const v16bf*>(p);
}

__device__ __forceinline__ v8f wmma_bf16(v16bf a, v16bf b, v8f c) {
    return __builtin_amdgcn_wmma_f32_16x16x32_bf16(false, a, false, b,
                                                   (short)0, c, false, false);
}

// ---------------------------------------------------------------------------
// Kernel 0: gather pruned mention rows into padded f32 [2000][1216]
// ---------------------------------------------------------------------------
__global__ void k_gather(const float* __restrict__ mr, const int* __restrict__ pidx,
                         float* __restrict__ G) {
    int p = blockIdx.x;
    long long idx = (long long)pidx[p];
    const float* src = mr + idx * GDIM;
    float* dst = G + (long long)p * GPAD;
    for (int c = threadIdx.x * 4; c < GPAD; c += blockDim.x * 4) {
        float4 v;
        if (c < GDIM) v = *reinterpret_cast<const float4*>(src + c);
        else          v = make_float4(0.f, 0.f, 0.f, 0.f);
        *reinterpret_cast<float4*>(dst + c) = v;
    }
}

// ---------------------------------------------------------------------------
// Kernel 1: convert weights to bf16, transposed B-layouts (padded with zeros)
//   W1abT[320][1216]: cols 0..149 <- W1a (rows 0..1199 of W1), 160..309 <- W1b
//   W1cT [160][1216]: cols 0..149 <- W1c (rows 2400..3599 of W1)
//   W2T  [160][160]
// ---------------------------------------------------------------------------
__global__ void k_wconv(const float* __restrict__ W1, const float* __restrict__ W2,
                        unsigned short* __restrict__ W1abT,
                        unsigned short* __restrict__ W1cT,
                        unsigned short* __restrict__ W2T) {
    const int N1 = 320 * GPAD, N2 = HPAD * GPAD, N3 = HPAD * HPAD;
    int id = blockIdx.x * blockDim.x + threadIdx.x;
    if (id < N1) {
        int n = id / GPAD, k = id - n * GPAD;
        float v = 0.f;
        if (k < GDIM) {
            if (n < HID)                      v = W1[k * HID + n];
            else if (n >= HPAD && n < HPAD + HID) v = W1[(GDIM + k) * HID + (n - HPAD)];
        }
        W1abT[id] = f2bf(v);
    } else if (id < N1 + N2) {
        int t = id - N1;
        int n = t / GPAD, k = t - n * GPAD;
        float v = (n < HID && k < GDIM) ? W1[(2 * GDIM + k) * HID + n] : 0.f;
        W1cT[t] = f2bf(v);
    } else if (id < N1 + N2 + N3) {
        int t = id - N1 - N2;
        int n = t / HPAD, k = t - n * HPAD;
        float v = (n < HID && k < HID) ? W2[k * HID + n] : 0.f;
        W2T[t] = f2bf(v);
    }
}

// ---------------------------------------------------------------------------
// Kernel 2: AB = G_bf16 @ [W1a|W1b]  -> f32 [2000][320]
// grid (32, 2), 256 thr: block computes 64 rows x 160 cols; 8 waves x 5 tiles.
// ---------------------------------------------------------------------------
__global__ void k_ab(const float* __restrict__ G,
                     const unsigned short* __restrict__ W1abT,
                     float* __restrict__ AB) {
    __shared__ __align__(16) unsigned short S[64 * 136];
    int tid = threadIdx.x, lane = tid & 31, wave = tid >> 5;
    int rowtile = (wave & 3) * 16, colgrp = wave >> 2;
    int rb = blockIdx.x * 64;
    int nb = blockIdx.y * HPAD;

    v8f acc[5];
#pragma unroll
    for (int t = 0; t < 5; ++t) acc[t] = (v8f){0.f, 0.f, 0.f, 0.f, 0.f, 0.f, 0.f, 0.f};

    for (int kb = 0; kb < GPAD; kb += 128) {
        int kc = imin(128, GPAD - kb);
        // stage bf16 tile of G rows into LDS
        for (int e = tid * 4; e < 64 * kc; e += 1024) {
            int r = e / kc, c = e - r * kc;
            int grow = rb + r;
            float4 v = make_float4(0.f, 0.f, 0.f, 0.f);
            if (grow < PNUM) v = *reinterpret_cast<const float4*>(G + (long long)grow * GPAD + kb + c);
            uint2 u;
            u.x = (unsigned)f2bf(v.x) | ((unsigned)f2bf(v.y) << 16);
            u.y = (unsigned)f2bf(v.z) | ((unsigned)f2bf(v.w) << 16);
            *reinterpret_cast<uint2*>(&S[r * 136 + c]) = u;
        }
        __syncthreads();
        for (int k0 = 0; k0 < kc; k0 += 32) {
            v16bf a = load_a_lds(S, 136, rowtile, k0, lane);
#pragma unroll
            for (int t = 0; t < 5; ++t) {
                int col = nb + (colgrp * 5 + t) * 16 + (lane & 15);
                v16bf b = load_b_glb(W1abT, GPAD, col, kb + k0, lane);
                acc[t] = wmma_bf16(a, b, acc[t]);
            }
        }
        __syncthreads();
    }
#pragma unroll
    for (int t = 0; t < 5; ++t) {
        int col = nb + (colgrp * 5 + t) * 16 + (lane & 15);
#pragma unroll
        for (int r = 0; r < 8; ++r) {
            int row = rowtile + r + (lane >> 4) * 8;
            int grow = rb + row;
            if (grow < PNUM) AB[grow * 320 + col] = acc[t][r];
        }
    }
}

// ---------------------------------------------------------------------------
// Kernel 3: fused per-anchor pair scoring. blockIdx.x = i in [0,2000).
//   S = (g_i ∘ G_window) bf16 ; H = S @ W1c (WMMA)
//   h1 = relu(H + A_i + B_j + b1) ; h2 = relu(h1 @ W2 + b2) (WMMA)
//   out = h2 @ W3 + b3 (VALU dot)
// ---------------------------------------------------------------------------
__global__ void k_pair(const float* __restrict__ G,
                       const unsigned short* __restrict__ W1cT,
                       const unsigned short* __restrict__ W2T,
                       const float* __restrict__ AB,
                       const float* __restrict__ b1, const float* __restrict__ b2,
                       const float* __restrict__ W3, const float* __restrict__ b3,
                       float* __restrict__ out) {
    __shared__ __align__(16) unsigned char smemA[64 * 152 * 4]; // S (bf16) / h2 (f32)
    __shared__ __align__(16) unsigned short h1[64 * 168];
    __shared__ __align__(16) float gi[GPAD];
    unsigned short* S = reinterpret_cast<unsigned short*>(smemA);
    float* h2 = reinterpret_cast<float*>(smemA);

    int i = blockIdx.x;
    int cnt = imin(i, KWIN);
    int tid = threadIdx.x, lane = tid & 31, wave = tid >> 5;
    int rowtile = (wave & 3) * 16, colgrp = wave >> 2;

    // stage g_i (f32) into LDS
    for (int c = tid * 4; c < GPAD; c += 1024)
        *reinterpret_cast<float4*>(gi + c) =
            *reinterpret_cast<const float4*>(G + (long long)i * GPAD + c);
    __syncthreads();

    v8f acc[5];
#pragma unroll
    for (int t = 0; t < 5; ++t) acc[t] = (v8f){0.f, 0.f, 0.f, 0.f, 0.f, 0.f, 0.f, 0.f};

    // GEMM1: bilinear term, K = 1216 in chunks of 128
    for (int kb = 0; kb < GPAD; kb += 128) {
        int kc = imin(128, GPAD - kb);
        for (int e = tid * 4; e < 64 * kc; e += 1024) {
            int r = e / kc, c = e - r * kc;
            float4 v = make_float4(0.f, 0.f, 0.f, 0.f);
            if (r < cnt) {
                int j = i - cnt + r;
                float4 gj = *reinterpret_cast<const float4*>(G + (long long)j * GPAD + kb + c);
                float4 g4 = *reinterpret_cast<const float4*>(gi + kb + c);
                v.x = gj.x * g4.x; v.y = gj.y * g4.y;
                v.z = gj.z * g4.z; v.w = gj.w * g4.w;
            }
            uint2 u;
            u.x = (unsigned)f2bf(v.x) | ((unsigned)f2bf(v.y) << 16);
            u.y = (unsigned)f2bf(v.z) | ((unsigned)f2bf(v.w) << 16);
            *reinterpret_cast<uint2*>(&S[r * 136 + c]) = u;
        }
        __syncthreads();
        for (int k0 = 0; k0 < kc; k0 += 32) {
            v16bf a = load_a_lds(S, 136, rowtile, k0, lane);
#pragma unroll
            for (int t = 0; t < 5; ++t) {
                int col = (colgrp * 5 + t) * 16 + (lane & 15);
                v16bf b = load_b_glb(W1cT, GPAD, col, kb + k0, lane);
                acc[t] = wmma_bf16(a, b, acc[t]);
            }
        }
        __syncthreads();
    }

    // epilogue 1: h1 = relu(H + A_i + B_j + b1) -> bf16 LDS [64][168]
    const float* ABi = AB + i * 320;
#pragma unroll
    for (int t = 0; t < 5; ++t) {
        int n = (colgrp * 5 + t) * 16 + (lane & 15);
        float ai  = ABi[n];
        float b1v = (n < HID) ? b1[n] : 0.f;
#pragma unroll
        for (int r = 0; r < 8; ++r) {
            int row = rowtile + r + (lane >> 4) * 8;
            float h = acc[t][r] + ai + b1v;
            if (row < cnt) h += AB[(i - cnt + row) * 320 + HPAD + n];
            h = fmaxf(h, 0.f);
            h1[row * 168 + n] = f2bf(h);
        }
    }
    __syncthreads();

    // GEMM2: h2 = h1 @ W2, K = 160
    v8f acc2[5];
#pragma unroll
    for (int t = 0; t < 5; ++t) acc2[t] = (v8f){0.f, 0.f, 0.f, 0.f, 0.f, 0.f, 0.f, 0.f};
    for (int k0 = 0; k0 < HPAD; k0 += 32) {
        v16bf a = load_a_lds(h1, 168, rowtile, k0, lane);
#pragma unroll
        for (int t = 0; t < 5; ++t) {
            int col = (colgrp * 5 + t) * 16 + (lane & 15);
            v16bf b = load_b_glb(W2T, HPAD, col, k0, lane);
            acc2[t] = wmma_bf16(a, b, acc2[t]);
        }
    }

    // epilogue 2: relu(+b2) -> f32 LDS [64][152] (only cols < 150 needed)
#pragma unroll
    for (int t = 0; t < 5; ++t) {
        int n = (colgrp * 5 + t) * 16 + (lane & 15);
        float b2v = (n < HID) ? b2[n] : 0.f;
#pragma unroll
        for (int r = 0; r < 8; ++r) {
            int row = rowtile + r + (lane >> 4) * 8;
            if (n < HID) h2[row * 152 + n] = fmaxf(acc2[t][r] + b2v, 0.f);
        }
    }
    __syncthreads();

    // final: out[row] = h2[row] . W3 + b3
    if (tid < cnt) {
        float s = b3[0];
        for (int c = 0; c < HID; ++c) s = fmaf(h2[tid * 152 + c], W3[c], s);
        int base = (i <= KWIN) ? (i * (i - 1)) / 2 : 1225 + (i - KWIN) * KWIN;
        out[base + tid] = s;
    }
}

// ---------------------------------------------------------------------------
extern "C" void kernel_launch(void* const* d_in, const int* in_sizes, int n_in,
                              void* d_out, int out_size, void* d_ws, size_t ws_size,
                              hipStream_t stream) {
    const float* mr   = (const float*)d_in[0];
    const int*   pidx = (const int*)d_in[1];
    // d_in[2] = K (50, static)
    const float* W1 = (const float*)d_in[3];
    const float* b1 = (const float*)d_in[4];
    const float* W2 = (const float*)d_in[5];
    const float* b2 = (const float*)d_in[6];
    const float* W3 = (const float*)d_in[7];
    const float* b3 = (const float*)d_in[8];
    float* out = (float*)d_out;

    char* ws = (char*)d_ws;
    float*          G     = (float*)(ws + OFF_G);
    float*          AB    = (float*)(ws + OFF_AB);
    unsigned short* W1abT = (unsigned short*)(ws + OFF_W1ABT);
    unsigned short* W1cT  = (unsigned short*)(ws + OFF_W1CT);
    unsigned short* W2T   = (unsigned short*)(ws + OFF_W2T);

    k_gather<<<PNUM, 256, 0, stream>>>(mr, pidx, G);

    const int wtot = 320 * GPAD + HPAD * GPAD + HPAD * HPAD; // 609280
    k_wconv<<<(wtot + 255) / 256, 256, 0, stream>>>(W1, W2, W1abT, W1cT, W2T);

    dim3 gab((PNUM + 63) / 64, 2);
    k_ab<<<gab, 256, 0, stream>>>(G, W1abT, AB);

    k_pair<<<PNUM, 256, 0, stream>>>(G, W1cT, W2T, AB, b1, b2, W3, b3, out);
}
